// AdaptiveTimeFrequency_64055142252568
// MI455X (gfx1250) — matrix operand order repaired and checked by reading, not compiled, not run on hardware
//
#include <hip/hip_runtime.h>
#include <cstdint>

#define B_ 256
#define F_ 32
#define L_ 4096

typedef __attribute__((ext_vector_type(16))) _Float16 v16h;
typedef __attribute__((ext_vector_type(8)))  _Float16 v8h;
typedef __attribute__((ext_vector_type(4)))  _Float16 v4h;
typedef __attribute__((ext_vector_type(8)))  float    v8f;
typedef __attribute__((ext_vector_type(4)))  int      v4i;

// ---------------------------------------------------------------------------
// Kernel 1: g[f,m] = softmax(window[f,:])[m] * cos(2*pi*(0.5*sigmoid(fr[f]))*m)
// stored as f16. One block per filter row.
// ---------------------------------------------------------------------------
__global__ __launch_bounds__(256)
void atf_prep_kernel(const float* __restrict__ wp,
                     const float* __restrict__ fp,
                     _Float16* __restrict__ gh) {
  __shared__ float red[256];
  const int f   = blockIdx.x;
  const int tid = threadIdx.x;
  const float* wrow = wp + (size_t)f * L_;

  float mx = -3.4e38f;
  for (int i = tid; i < L_; i += 256) mx = fmaxf(mx, wrow[i]);
  red[tid] = mx;
  __syncthreads();
  for (int s = 128; s > 0; s >>= 1) {
    if (tid < s) red[tid] = fmaxf(red[tid], red[tid + s]);
    __syncthreads();
  }
  mx = red[0];
  __syncthreads();

  float sum = 0.f;
  for (int i = tid; i < L_; i += 256) sum += __expf(wrow[i] - mx);
  red[tid] = sum;
  __syncthreads();
  for (int s = 128; s > 0; s >>= 1) {
    if (tid < s) red[tid] += red[tid + s];
    __syncthreads();
  }
  sum = red[0];

  const float inv  = 1.0f / sum;
  const float freq = 0.5f / (1.0f + __expf(-fp[f]));
  const float w2pi = 6.28318530717958647692f * freq;

  for (int i = tid; i < L_; i += 256) {
    float g = __expf(wrow[i] - mx) * inv * __cosf(w2pi * (float)i);
    gh[(size_t)f * L_ + i] = (_Float16)g;
  }
}

// ---------------------------------------------------------------------------
// Kernel 2: convert x to f16 once (so kernel 3 can async-DMA it into LDS).
// ---------------------------------------------------------------------------
__global__ __launch_bounds__(256)
void atf_cvt_kernel(const float* __restrict__ x, _Float16* __restrict__ xh) {
  const size_t i = ((size_t)blockIdx.x * 256 + threadIdx.x) * 4;
  float4 v = *(const float4*)(x + i);
  v4h h;
  h[0] = (_Float16)v.x; h[1] = (_Float16)v.y;
  h[2] = (_Float16)v.z; h[3] = (_Float16)v.w;
  *(v4h*)(xh + i) = h;
}

// ---------------------------------------------------------------------------
// Kernel 3: out[b,f,t] = sum_m g[f,m] * x[b,(t+m) mod L] via WMMA.
// grid = (L/128, B); 8 waves; wave = one 16-wide t tile x both 16-row f tiles.
// LDS: xe = duplicated row (2L halves), xo = half-shifted dword-packed copy
// (+16-dword skew for bank spread) -> every lane's B fragment is 8 aligned
// ds_load_b32, no realignment VALU.
// ---------------------------------------------------------------------------
__global__ __launch_bounds__(256)
void atf_corr_kernel(const _Float16* __restrict__ xh,
                     const _Float16* __restrict__ gh,
                     float* __restrict__ out) {
  __shared__ uint32_t shbuf[4096 + 16 + 4096];   // xe | pad | xo  (~32 KB)
  const int tid = threadIdx.x;
  const int b   = blockIdx.y;

  _Float16*       xe   = (_Float16*)shbuf;
  uint32_t*       xo   = shbuf + 4096 + 16;
  const _Float16* xrow = xh + (size_t)b * L_;

  // ---- stage xe = x row (f16), duplicated for circular indexing -----------
#if __has_builtin(__builtin_amdgcn_global_load_async_to_lds_b128)
  for (int c = tid; c < 512; c += 256) {
    __builtin_amdgcn_global_load_async_to_lds_b128(
        (__attribute__((address_space(1))) v4i*)(xrow + (size_t)c * 8),
        (__attribute__((address_space(3))) v4i*)(xe + c * 8), 0, 0);
    __builtin_amdgcn_global_load_async_to_lds_b128(
        (__attribute__((address_space(1))) v4i*)(xrow + (size_t)c * 8),
        (__attribute__((address_space(3))) v4i*)(xe + L_ + c * 8), 0, 0);
  }
  asm volatile("s_wait_asynccnt 0x0" ::: "memory");
#else
  for (int c = tid; c < 512; c += 256) {
    v4i v = ((const v4i*)xrow)[c];
    ((v4i*)xe)[c]       = v;
    ((v4i*)xe)[c + 512] = v;
  }
#endif
  __syncthreads();

  // ---- build odd-shifted copy: xo[j] = halves (2j+1, 2j+2) ----------------
  {
    const uint32_t* xe32 = shbuf;
    for (int j = tid * 4; j < 4096; j += 1024) {
      uint32_t w0 = xe32[j],     w1 = xe32[j + 1], w2 = xe32[j + 2];
      uint32_t w3 = xe32[j + 3], w4 = xe32[j + 4];           // j+4==4096 -> pad
      v4i o;
      o[0] = (int)((w0 >> 16) | (w1 << 16));
      o[1] = (int)((w1 >> 16) | (w2 << 16));
      o[2] = (int)((w2 >> 16) | (w3 << 16));
      o[3] = (int)((w3 >> 16) | (w4 << 16));
      *(v4i*)(xo + j) = o;
    }
  }
  __syncthreads();

  // ---- main GEMM loop -----------------------------------------------------
  const int lane = tid & 31;
  const int w    = tid >> 5;
  const int t0   = (blockIdx.x << 7) + (w << 4);   // 8 t-tiles of 16 per block
  const int n    = lane & 15;
  const int hi   = lane >> 4;

  // A layout (16-bit A 16x32): lanes 0-15 take K 0-15,
  // lanes 16-31 take K 16-31 (contiguous, matches B).
  const _Float16* a0row = gh + (size_t)n * L_ + (hi << 4);          // f tile 0
  const _Float16* a1row = a0row + (size_t)16 * L_;                  // f tile 1

  // B: lane n/half hi needs halves [s, s+16), s = t0 + n + hi*16 + m0.
  // Parity of s is loop-invariant -> pick even or odd LDS image once.
  const int       sbase = t0 + n + (hi << 4);
  const uint32_t* xe32  = shbuf;
  const uint32_t* bbase = (sbase & 1) ? (xo + ((sbase - 1) >> 1))
                                      : (xe32 + (sbase >> 1));

  v8f acc0 = {}, acc1 = {};

#pragma unroll 2
  for (int m0 = 0; m0 < L_; m0 += 32) {
    const v8h* a0p = (const v8h*)(a0row + m0);
    const v8h* a1p = (const v8h*)(a1row + m0);
    v8h a00 = a0p[0], a01 = a0p[1];
    v8h a10 = a1p[0], a11 = a1p[1];

    union { uint32_t u[8]; v16h h; } bu;
    const uint32_t* bp = bbase + (m0 >> 1);
#pragma unroll
    for (int i = 0; i < 8; i++) bu.u[i] = bp[i];

    v16h A0, A1;
#pragma unroll
    for (int i = 0; i < 8; i++) {
      A0[i] = a00[i]; A0[i + 8] = a01[i];
      A1[i] = a10[i]; A1[i + 8] = a11[i];
    }

    acc0 = __builtin_amdgcn_wmma_f32_16x16x32_f16(false, A0, false, bu.h,
                                                  (short)0, acc0, false, false);
    acc1 = __builtin_amdgcn_wmma_f32_16x16x32_f16(false, A1, false, bu.h,
                                                  (short)0, acc1, false, false);
  }

  // C/D layout: VGPR r -> (M = r + hi*8, N = n); M = filter, N = time.
  float* op = out + ((size_t)b * F_ + (size_t)(hi << 3)) * (size_t)L_
                  + (size_t)(t0 + n);
#pragma unroll
  for (int r = 0; r < 8; r++) {
    op[(size_t)r * L_]        = acc0[r];   // f = hi*8 + r
    op[(size_t)(16 + r) * L_] = acc1[r];   // f = 16 + hi*8 + r
  }
}

// ---------------------------------------------------------------------------
extern "C" void kernel_launch(void* const* d_in, const int* in_sizes, int n_in,
                              void* d_out, int out_size, void* d_ws, size_t ws_size,
                              hipStream_t stream) {
  const float* x  = (const float*)d_in[0];   // [B, L] f32
  const float* wp = (const float*)d_in[1];   // [F, L] f32
  const float* fp = (const float*)d_in[2];   // [1, F] f32
  float* out      = (float*)d_out;           // [B, F, L] f32

  _Float16* gh  = (_Float16*)d_ws;                                 // 256 KB
  _Float16* xhp = (_Float16*)((char*)d_ws + (size_t)F_ * L_ * 2);  // 2 MB

  atf_prep_kernel<<<dim3(F_), dim3(256), 0, stream>>>(wp, fp, gh);
  atf_cvt_kernel<<<dim3((B_ * L_) / 1024), dim3(256), 0, stream>>>(x, xhp);
  atf_corr_kernel<<<dim3(L_ / 128, B_), dim3(256), 0, stream>>>(xhp, gh, out);
}
